// MBT_919123002043
// MI455X (gfx1250) — matrix operand order repaired and checked by reading, not compiled, not run on hardware
//
#include <hip/hip_runtime.h>
#include <hip/hip_bf16.h>
#include <math.h>

// ---------------------------------------------------------------------------
// Fused cosine-sim retrieval on gfx1250:
//   k0: normalize queries -> bf16
//   k1: convert memory bank -> bf16 (streaming, once per launch)
//   k2: streaming bf16-WMMA GEMM + running top-5 (2 A-tiles per wave,
//       full B tile preloaded before the 8-WMMA chain)
//   k3: merge partials, softmax(T=0.1), weighted gather, renormalize
// Workspace: ~53 MB (bf16 bank 51.2 MB + bf16 queries + partial top-k).
// ---------------------------------------------------------------------------

typedef __attribute__((ext_vector_type(16))) __bf16        v16bf;
typedef __attribute__((ext_vector_type(8)))  float         v8f;
typedef __attribute__((ext_vector_type(4)))  float         f32x4;
typedef __attribute__((ext_vector_type(4)))  unsigned int  u32x4;
typedef __attribute__((ext_vector_type(8)))  unsigned int  u32x8;
typedef __attribute__((ext_vector_type(2)))  unsigned int  u32x2;
typedef __attribute__((ext_vector_type(4)))  __bf16        bf16x4;
typedef __attribute__((ext_vector_type(8)))  __bf16        bf16x8;

#define DDIM   128
#define TOPK   5
#define SEGS   32
#define WAVES  4
#define MROWS  32      /* queries per block in the GEMM kernel (2 WMMA M-tiles) */
#define INVT   10.0f   /* 1 / TEMPERATURE */

__device__ __attribute__((always_inline)) inline
void insert5(float v, int n, float (&tv)[TOPK], int (&tn)[TOPK]) {
    if (v <= tv[4]) return;
    tv[4] = v; tn[4] = n;
#pragma unroll
    for (int j = 4; j > 0; --j) {
        if (tv[j] > tv[j - 1]) {
            float t = tv[j]; tv[j] = tv[j - 1]; tv[j - 1] = t;
            int   u = tn[j]; tn[j] = tn[j - 1]; tn[j - 1] = u;
        }
    }
}

// ---------------------------------------------------------------------------
// Kernel 0: normalize queries (x / max(||x||, 1e-12)), convert to bf16.
// One wave per query row.
// ---------------------------------------------------------------------------
__global__ __launch_bounds__(256)
void MBT_qnorm_kernel(const float* __restrict__ q,
                      unsigned short* __restrict__ qn, int B) {
    const int wave = threadIdx.x >> 5;
    const int lane = threadIdx.x & 31;
    const int row  = blockIdx.x * 8 + wave;
    if (row >= B) return;

    const f32x4* qp = (const f32x4*)(q + (size_t)row * DDIM);
    f32x4 v = qp[lane];
    float ss = v[0]*v[0] + v[1]*v[1] + v[2]*v[2] + v[3]*v[3];
#pragma unroll
    for (int off = 16; off > 0; off >>= 1) ss += __shfl_xor(ss, off, 32);
    float scale = 1.0f / fmaxf(sqrtf(ss), 1e-12f);

    bf16x4 o;
    o[0] = (__bf16)(v[0] * scale);
    o[1] = (__bf16)(v[1] * scale);
    o[2] = (__bf16)(v[2] * scale);
    o[3] = (__bf16)(v[3] * scale);
    *(u32x2*)(qn + (size_t)row * DDIM + lane * 4) = __builtin_bit_cast(u32x2, o);
}

// ---------------------------------------------------------------------------
// Kernel 1: convert the (already normalized) memory bank f32 -> bf16.
// 8 elements per thread, 16B stores.
// ---------------------------------------------------------------------------
__global__ __launch_bounds__(256)
void MBT_bank_kernel(const float* __restrict__ mem,
                     unsigned short* __restrict__ membf, long long total8) {
    const long long i = (long long)blockIdx.x * 256 + threadIdx.x;
    if (i >= total8) return;
    const f32x4* src = (const f32x4*)mem + i * 2;
    f32x4 a = src[0], b = src[1];
    bf16x8 o;
    o[0] = (__bf16)a[0]; o[1] = (__bf16)a[1]; o[2] = (__bf16)a[2]; o[3] = (__bf16)a[3];
    o[4] = (__bf16)b[0]; o[5] = (__bf16)b[1]; o[6] = (__bf16)b[2]; o[7] = (__bf16)b[3];
    ((u32x4*)membf)[i] = __builtin_bit_cast(u32x4, o);
}

// ---------------------------------------------------------------------------
// Kernel 2: per (32-query block, N-segment) streaming WMMA + running top-5.
// block = 128 threads = 4 waves; each wave holds TWO 16x128 A-tiles in
// registers, preloads the full 16x128 bf16 B tile (8 x b128), then runs
// 8 chained WMMAs.
// ---------------------------------------------------------------------------
__global__ __launch_bounds__(128)
void MBT_topk_partial_kernel(const unsigned short* __restrict__ qn,
                             const unsigned short* __restrict__ membf,
                             const long long* __restrict__ imap,
                             const long long* __restrict__ excl,
                             float* __restrict__ pV, int* __restrict__ pI,
                             int N, int segLen, int nSegs) {
    __shared__ float     simLds[WAVES][2][16][17];
    __shared__ long long idxLds[WAVES][16];
    __shared__ float     mV[WAVES * 2][MROWS][TOPK];
    __shared__ int       mN[WAVES * 2][MROWS][TOPK];

    const int wave = threadIdx.x >> 5;
    const int lane = threadIdx.x & 31;
    const int half = lane >> 4;     // which K-half of the fragment this lane holds
    const int lrow = lane & 15;
    const int m0   = blockIdx.x * MROWS;
    const int segStart = blockIdx.y * segLen;
    const int segEnd   = min(segStart + segLen, N);

    const long long excl0 = excl[m0 + lrow];
    const long long excl1 = excl[m0 + 16 + lrow];

    // --- Preload A fragments for both M-tiles (ISA 16-bit A layout) ---
    // lane<16 : row=lrow, K = {k0..k0+7, k0+16..k0+23}
    // lane>=16: row=lrow, K = {k0+8..k0+15, k0+24..k0+31}
    v16bf afr0[4], afr1[4];
    {
        const u32x4* qp = (const u32x4*)qn;               // 8 bf16 per u32x4
        const int rb0 = (m0 + lrow) * 16;                 // row stride: 16 u32x4
        const int rb1 = (m0 + 16 + lrow) * 16;
#pragma unroll
        for (int kk = 0; kk < 4; ++kk) {
            u32x4 lo = qp[rb0 + kk * 4 + half];
            u32x4 hi = qp[rb0 + kk * 4 + half + 2];
            u32x8 c; c[0]=lo[0]; c[1]=lo[1]; c[2]=lo[2]; c[3]=lo[3];
                     c[4]=hi[0]; c[5]=hi[1]; c[6]=hi[2]; c[7]=hi[3];
            afr0[kk] = __builtin_bit_cast(v16bf, c);
            lo = qp[rb1 + kk * 4 + half];
            hi = qp[rb1 + kk * 4 + half + 2];
            u32x8 d; d[0]=lo[0]; d[1]=lo[1]; d[2]=lo[2]; d[3]=lo[3];
                     d[4]=hi[0]; d[5]=hi[1]; d[6]=hi[2]; d[7]=hi[3];
            afr1[kk] = __builtin_bit_cast(v16bf, d);
        }
    }

    float tv0[TOPK], tv1[TOPK]; int tn0[TOPK], tn1[TOPK];
#pragma unroll
    for (int k = 0; k < TOPK; ++k) {
        tv0[k] = -__builtin_inff(); tn0[k] = -1;
        tv1[k] = -__builtin_inff(); tn1[k] = -1;
    }

    const int nTiles = (segEnd - segStart + 15) >> 4;
    for (int t = wave; t < nTiles; t += WAVES) {
        const int n0 = segStart + (t << 4);

        if (lane < 16)
            idxLds[wave][lane] = imap[min(n0 + lane, N - 1)];

        // --- preload full bf16 B tile for this lane's column + K-half:
        //     8 x b128 in one clause, distinct registers ---
        const int n = min(n0 + lrow, N - 1);
        const u32x4* bp = (const u32x4*)membf + (size_t)n * 16;   // 16 u32x4/row
        u32x4 bl[8];
#pragma unroll
        for (int kk = 0; kk < 4; ++kk) {
            bl[2 * kk + 0] = bp[kk * 4 + half];
            bl[2 * kk + 1] = bp[kk * 4 + half + 2];
        }

        v8f c0 = {0.f,0.f,0.f,0.f,0.f,0.f,0.f,0.f};
        v8f c1 = {0.f,0.f,0.f,0.f,0.f,0.f,0.f,0.f};
#pragma unroll
        for (int kk = 0; kk < 4; ++kk) {
            u32x8 cb;
            cb[0] = bl[2*kk][0]; cb[1] = bl[2*kk][1];
            cb[2] = bl[2*kk][2]; cb[3] = bl[2*kk][3];
            cb[4] = bl[2*kk+1][0]; cb[5] = bl[2*kk+1][1];
            cb[6] = bl[2*kk+1][2]; cb[7] = bl[2*kk+1][3];
            v16bf bfr = __builtin_bit_cast(v16bf, cb);
            c0 = __builtin_amdgcn_wmma_f32_16x16x32_bf16(
                     false, afr0[kk], false, bfr, (short)0, c0, false, false);
            c1 = __builtin_amdgcn_wmma_f32_16x16x32_bf16(
                     false, afr1[kk], false, bfr, (short)0, c1, false, false);
        }

        // --- spill both C tiles to per-wave LDS (row = M, col = N) ---
#pragma unroll
        for (int r = 0; r < 8; ++r) {
            simLds[wave][0][r + half * 8][lrow] = c0[r];
            simLds[wave][1][r + half * 8][lrow] = c1[r];
        }

        // --- running top-5: lane covers 2 rows x 8 columns ---
#pragma unroll
        for (int j = 0; j < 8; ++j) {
            const int cidx = half * 8 + j;
            const int gn   = n0 + cidx;
            if (gn < segEnd) {
                const long long iv = idxLds[wave][cidx];
                float v0 = simLds[wave][0][lrow][cidx];
                if (iv != excl0) insert5(v0, gn, tv0, tn0);
                float v1 = simLds[wave][1][lrow][cidx];
                if (iv != excl1) insert5(v1, gn, tv1, tn1);
            }
        }
    }

    // --- publish per-lane lists (8 lists per row), merge in wave 0 ---
#pragma unroll
    for (int k = 0; k < TOPK; ++k) {
        mV[wave * 2 + half][lrow][k]      = tv0[k];
        mN[wave * 2 + half][lrow][k]      = tn0[k];
        mV[wave * 2 + half][16 + lrow][k] = tv1[k];
        mN[wave * 2 + half][16 + lrow][k] = tn1[k];
    }
    __syncthreads();

    if (wave == 0) {                       // 32 lanes, one row each
        float fv[TOPK]; int fn[TOPK];
#pragma unroll
        for (int k = 0; k < TOPK; ++k) { fv[k] = -__builtin_inff(); fn[k] = -1; }
        for (int l = 0; l < WAVES * 2; ++l)
#pragma unroll
            for (int k = 0; k < TOPK; ++k)
                insert5(mV[l][lane][k], mN[l][lane][k], fv, fn);

        const int b = m0 + lane;
        const size_t base = ((size_t)b * nSegs + blockIdx.y) * TOPK;
#pragma unroll
        for (int k = 0; k < TOPK; ++k) { pV[base + k] = fv[k]; pI[base + k] = fn[k]; }
    }
}

// ---------------------------------------------------------------------------
// Kernel 3: merge partials, softmax, weighted gather (f32 bank), renormalize.
// One wave per query; all lanes redundantly merge (uniform broadcast loads).
// ---------------------------------------------------------------------------
__global__ __launch_bounds__(256)
void MBT_finalize_kernel(const float* __restrict__ pV, const int* __restrict__ pI,
                         const float* __restrict__ mem, float* __restrict__ out,
                         int B, int nSegs) {
    const int wave = threadIdx.x >> 5;
    const int lane = threadIdx.x & 31;
    const int b = blockIdx.x * 8 + wave;
    if (b >= B) return;

    float tv[TOPK]; int tn[TOPK];
#pragma unroll
    for (int k = 0; k < TOPK; ++k) { tv[k] = -__builtin_inff(); tn[k] = -1; }

    const float* v = pV + (size_t)b * nSegs * TOPK;
    const int*   n = pI + (size_t)b * nSegs * TOPK;
    for (int i = 0; i < nSegs * TOPK; ++i) insert5(v[i], n[i], tv, tn);

    float e[TOPK], s = 0.f;
#pragma unroll
    for (int k = 0; k < TOPK; ++k) { e[k] = __expf((tv[k] - tv[0]) * INVT); s += e[k]; }
    const float inv = 1.0f / s;
    float w[TOPK];
#pragma unroll
    for (int k = 0; k < TOPK; ++k) w[k] = e[k] * inv;

    f32x4 acc = {0.f, 0.f, 0.f, 0.f};
#pragma unroll
    for (int k = 0; k < TOPK; ++k) {
        const int nk = tn[k] < 0 ? 0 : tn[k];
        const f32x4* mp = (const f32x4*)(mem + (size_t)nk * DDIM);
        f32x4 mv = mp[lane];
        acc[0] += w[k] * mv[0]; acc[1] += w[k] * mv[1];
        acc[2] += w[k] * mv[2]; acc[3] += w[k] * mv[3];
    }
    float ss = acc[0]*acc[0] + acc[1]*acc[1] + acc[2]*acc[2] + acc[3]*acc[3];
#pragma unroll
    for (int off = 16; off > 0; off >>= 1) ss += __shfl_xor(ss, off, 32);
    const float scale = 1.0f / fmaxf(sqrtf(ss), 1e-12f);

    f32x4 r;
    r[0] = acc[0]*scale; r[1] = acc[1]*scale; r[2] = acc[2]*scale; r[3] = acc[3]*scale;
    ((f32x4*)(out + (size_t)b * DDIM))[lane] = r;

    if (lane == 0) {
        out[(size_t)B * DDIM + b] = 1.0f - tv[0];                 // distances
        float* wout = out + (size_t)B * DDIM + B + (size_t)b * TOPK;
#pragma unroll
        for (int k = 0; k < TOPK; ++k) wout[k] = w[k];            // weights
    }
}

// ---------------------------------------------------------------------------
extern "C" void kernel_launch(void* const* d_in, const int* in_sizes, int n_in,
                              void* d_out, int out_size, void* d_ws, size_t ws_size,
                              hipStream_t stream) {
    const float*     query  = (const float*)d_in[0];
    const float*     memory = (const float*)d_in[1];
    const long long* imap   = (const long long*)d_in[2];
    const long long* excl   = (const long long*)d_in[3];
    float* out = (float*)d_out;

    const int B = in_sizes[0] / DDIM;     // 1024
    const int N = in_sizes[1] / DDIM;     // 200000
    const int segLen = (N + SEGS - 1) / SEGS;

    // ws layout: bf16 queries | partial vals | partial idx | bf16 bank
    char* ws = (char*)d_ws;
    size_t off = 0;
    unsigned short* qn = (unsigned short*)(ws + off);
    off += ((size_t)B * DDIM * 2 + 255) & ~(size_t)255;
    float* pV = (float*)(ws + off);
    off += (((size_t)B * SEGS * TOPK * 4) + 255) & ~(size_t)255;
    int* pI = (int*)(ws + off);
    off += (((size_t)B * SEGS * TOPK * 4) + 255) & ~(size_t)255;
    unsigned short* membf = (unsigned short*)(ws + off);   // N*128*2 bytes

    MBT_qnorm_kernel<<<(B + 7) / 8, 256, 0, stream>>>(query, qn, B);

    const long long total8 = (long long)N * DDIM / 8;
    MBT_bank_kernel<<<(int)((total8 + 255) / 256), 256, 0, stream>>>(memory, membf, total8);

    dim3 g1(B / MROWS, SEGS);
    MBT_topk_partial_kernel<<<g1, 128, 0, stream>>>(qn, membf, imap, excl,
                                                    pV, pI, N, segLen, SEGS);

    MBT_finalize_kernel<<<(B + 7) / 8, 256, 0, stream>>>(pV, pI, memory, out, B, SEGS);
}
